// KANLayer_70334384439343
// MI455X (gfx1250) — matrix operand order repaired and checked by reading, not compiled, not run on hardware
//
#include <hip/hip_runtime.h>

// ---------------------------------------------------------------------------
// KAN layer on gfx1250 via dense-expansion WMMA GEMM.
//   out[n,o] = sum_{i,k} wy[n,i,k]*coeffs[o,i,k] + wd[n,i,k]*d[o,i,k] + bias[o]
// K dimension = 128 dims * 64 knots * 2 slots = 16384, chunked 32 per WMMA.
// A (weights) built on the fly in registers; B (coeffs+slopes, f16) packed
// once per launch into d_ws in a contiguous-k B order matching A.
// ---------------------------------------------------------------------------

#define D_IN      128
#define D_OUT     256
#define N_SAMPLES 2048
#define NUM_KNOTS 64
#define X_MIN_F  (-2.0f)
#define X_MAX_F  ( 2.0f)
#define EPSF     (1e-12f)
#define HSTEP    (4.0f / 63.0f)

typedef __attribute__((ext_vector_type(16))) _Float16 v16h;
typedef __attribute__((ext_vector_type(8)))  float    v8f;
typedef __attribute__((ext_vector_type(4)))  unsigned v4u;

union Frag {
    v16h h;
    unsigned u[8];
    v4u  q[2];
};

__device__ __forceinline__ unsigned pack2h(float lo, float hi) {
    _Float16 l = (_Float16)lo;
    _Float16 h = (_Float16)hi;
    unsigned short ls = __builtin_bit_cast(unsigned short, l);
    unsigned short hs = __builtin_bit_cast(unsigned short, h);
    return (unsigned)ls | ((unsigned)hs << 16);
}

__device__ __forceinline__ float pchip_limit(float di, float da, float db) {
    di = (di * da <= 0.0f) ? 0.0f : di;
    di = ((da * db < 0.0f) && (fabsf(di) > 3.0f * fabsf(da))) ? 3.0f * da : di;
    return di;
}

// ---------------------------------------------------------------------------
// Pack B matrix: for chunk c (=4*i+kwin), o-tile ot, lane, vgpr v, emit the
// packed dword {lo: f16(coeffs[o,i,kk]), hi: f16(pchip_slope[o,i,kk])}.
// B order used here (16-bit B, 32x16, wave32; contiguous k, same as A): lane holds column n = lane&15;
// lanes 0-15 cover K=0..15, lanes 16-31 cover K=16..31; VGPR v holds K=2v,2v+1.
// So kk = kwin*16 + (lane>=16 ? 8 : 0) + v ; slot = K&1 (lo=value, hi=slope).
// tid enumerates (c, ot, lane, v) with v fastest -> wsB[tid].
// ---------------------------------------------------------------------------
__global__ __launch_bounds__(256) void pack_b_kernel(
    const float* __restrict__ coeffs, unsigned* __restrict__ wsB) {
    int tid  = blockIdx.x * blockDim.x + threadIdx.x;   // < 512*16*32*8 = 2097152
    int v    = tid & 7;
    int lane = (tid >> 3) & 31;
    int ot   = (tid >> 8) & 15;
    int c    = tid >> 12;                               // 0..511
    int i    = c >> 2;
    int kwin = c & 3;
    int kk   = (kwin << 4) + ((lane >> 4) << 3) + v;    // 0..63
    int o    = (ot << 4) + (lane & 15);

    const float* y = coeffs + (o * D_IN + i) * NUM_KNOTS;
    const float H = HSTEP;

    float ycur = y[kk];
    float d;
    if (kk == 0) {
        float d0 = (y[1] - y[0]) / (H + EPSF);
        float d1 = (y[2] - y[1]) / (H + EPSF);
        float dd = (3.0f * H * d0 - H * d1) / (2.0f * H + EPSF);
        d = pchip_limit(dd, d0, d1);
    } else if (kk == NUM_KNOTS - 1) {
        float dl = (y[63] - y[62]) / (H + EPSF);
        float dp = (y[62] - y[61]) / (H + EPSF);
        float dd = (3.0f * H * dl - H * dp) / (2.0f * H + EPSF);
        d = pchip_limit(dd, dl, dp);
    } else {
        float dm1 = (y[kk]     - y[kk - 1]) / (H + EPSF);
        float dk  = (y[kk + 1] - y[kk])     / (H + EPSF);
        float w   = 3.0f * H;                 // 2*hk+hkm1 == hk+2*hkm1 == 3h (uniform)
        float denom = w / (dm1 + EPSF) + w / (dk + EPSF);
        d = (dm1 * dk > 0.0f) ? (2.0f * w) / (denom + EPSF) : 0.0f;
    }
    wsB[tid] = pack2h(ycur, d);
}

// ---------------------------------------------------------------------------
// Main kernel: grid.x = 128 n-tiles (16 samples each). Block = 128 threads =
// 4 waves; wave w owns o-tiles 4w..4w+3 (32 f32 accumulator VGPRs) and walks
// the full K range. A fragment built per chunk from x, reused for 4 WMMAs.
// A layout (16-bit A, 16x32): lane holds row m = lane&15; VGPR v holds
// K = (lane>=16?16:0) + 2*v + {0,1} (matches packed B)  ->
// kk(v) = kwin*16 + (lane>=16?8:0) + v, both slots packed.
// ---------------------------------------------------------------------------
__global__ __launch_bounds__(128) void kan_wmma_kernel(
    const float* __restrict__ x, const float* __restrict__ bias,
    const unsigned* __restrict__ wsB, float* __restrict__ out) {
    const int lane = threadIdx.x & 31;
    const int wv   = threadIdx.x >> 5;         // wave in block: 0..3
    const int m    = lane & 15;                // A row / sample within tile
    const int hh4  = ((lane >> 4) & 1) << 3;   // A K-sub-offset for lane half
    const int n0   = blockIdx.x << 4;

    v8f zero = {0.f, 0.f, 0.f, 0.f, 0.f, 0.f, 0.f, 0.f};
    v8f acc[4];
    acc[0] = zero; acc[1] = zero; acc[2] = zero; acc[3] = zero;

    const float H = HSTEP;

    for (int i = 0; i < D_IN; ++i) {
        float xv = x[(n0 + m) * D_IN + i];

        // Hermite weights for this (sample, input-dim); extrapolation folds in.
        int idx;
        unsigned p0, p1;   // p0 -> knot idx {wy, wd}; p1 -> knot idx+1 {wy, wd}
        if (xv < X_MIN_F) {
            idx = 0;
            p0  = pack2h(1.0f, xv - X_MIN_F);
            p1  = 0u;
        } else if (xv > X_MAX_F) {
            idx = NUM_KNOTS - 2;               // weights live on knot idx+1 = 63
            p0  = 0u;
            p1  = pack2h(1.0f, xv - X_MAX_F);
        } else {
            float u = (xv - X_MIN_F) * (63.0f / 4.0f);
            int ii = (int)u;
            ii = ii < 0 ? 0 : (ii > NUM_KNOTS - 2 ? NUM_KNOTS - 2 : ii);
            idx = ii;
            float t  = (xv - (X_MIN_F + (float)ii * H)) / H;
            float t2 = t * t, t3 = t2 * t;
            float h00 =  2.0f * t3 - 3.0f * t2 + 1.0f;
            float h10 =  t3 - 2.0f * t2 + t;
            float h01 = -2.0f * t3 + 3.0f * t2;
            float h11 =  t3 - t2;
            p0 = pack2h(h00, h10 * H);
            p1 = pack2h(h01, h11 * H);
        }

#pragma unroll
        for (int kwin = 0; kwin < 4; ++kwin) {
            // Build sparse A fragment (2 nonzero dwords out of 8 across halves).
            Frag a;
#pragma unroll
            for (int v = 0; v < 8; ++v) {
                int kkv = (kwin << 4) + hh4 + v;
                a.u[v] = (kkv == idx) ? p0 : ((kkv == idx + 1) ? p1 : 0u);
            }

            int c  = (i << 2) + kwin;
            int cn = (c + 1 < 512) ? (c + 1) : c;
            // Prefetch next chunk's B for this wave's first o-tile.
            __builtin_prefetch(
                wsB + (((cn << 4) + (wv << 2)) << 8) + (lane << 3), 0, 1);

#pragma unroll
            for (int j = 0; j < 4; ++j) {
                int ot = (wv << 2) + j;
                const unsigned* bp = wsB + (((c << 4) + ot) << 8) + (lane << 3);
                Frag b;
                b.q[0] = *(const v4u*)bp;
                b.q[1] = *(const v4u*)(bp + 4);
                acc[j] = __builtin_amdgcn_wmma_f32_16x16x32_f16(
                    false, a.h, false, b.h, (short)0, acc[j], false, false);
            }
        }
    }

    // D layout (16x16 f32): VGPR r, lanes 0-15 -> M=r, lanes 16-31 -> M=r+8;
    // N = lane & 15.
#pragma unroll
    for (int j = 0; j < 4; ++j) {
        int o  = (((wv << 2) + j) << 4) + (lane & 15);
        float bv = bias[o];
#pragma unroll
        for (int r = 0; r < 8; ++r) {
            int mm = r + ((lane >> 4) << 3);
            out[(n0 + mm) * D_OUT + o] = acc[j][r] + bv;
        }
    }
}

extern "C" void kernel_launch(void* const* d_in, const int* in_sizes, int n_in,
                              void* d_out, int out_size, void* d_ws, size_t ws_size,
                              hipStream_t stream) {
    const float* x      = (const float*)d_in[0];   // (2048, 128)
    const float* coeffs = (const float*)d_in[1];   // (256, 128, 64)
    const float* bias   = (const float*)d_in[2];   // (256,)
    // d_in[3] = knots: uniform linspace(-2,2,64); folded into constants.
    float* out = (float*)d_out;                    // (2048, 256)

    unsigned* wsB = (unsigned*)d_ws;               // 512*16*32*8 dwords = 8 MB

    // 1) Pack coeffs + PCHIP slopes into fp16 WMMA B fragments.
    const int packN = 512 * 16 * 32 * 8;           // 2,097,152 dwords
    pack_b_kernel<<<packN / 256, 256, 0, stream>>>(coeffs, wsB);

    // 2) WMMA GEMM: 128 n-tile blocks x 4 waves, 2048 WMMAs per wave.
    kan_wmma_kernel<<<N_SAMPLES / 16, 128, 0, stream>>>(x, bias, wsB, out);
}
